// Groupwise_CNN_LSTM_Pr_61452392071760
// MI455X (gfx1250) — matrix operand order repaired and checked by reading, not compiled, not run on hardware
//
#include <hip/hip_runtime.h>
#include <hip/hip_bf16.h>

typedef _Float16 v16h __attribute__((ext_vector_type(16)));
typedef _Float16 v8h  __attribute__((ext_vector_type(8)));
typedef float    v8f  __attribute__((ext_vector_type(8)));

#define BATCH 128
#define WIN   128      // time steps / conv groups
#define NFEAT 128      // conv spatial
#define KC    16
#define POOL  65       // (128+2)/2
#define FC1N  32
#define HID   32
#define G4H   128      // 4*HID
#define FCK   1040     // KC*POOL
#define FCKP  1056     // padded to 33*32

__device__ __forceinline__ float sigf(float x)  { return 1.0f / (1.0f + __expf(-x)); }
__device__ __forceinline__ float tanhg(float x) { return 2.0f / (1.0f + __expf(-2.0f * x)) - 1.0f; }

__device__ __forceinline__ v16h cat8(v8h a, v8h b) {
    return __builtin_shufflevector(a, b, 0,1,2,3,4,5,6,7,8,9,10,11,12,13,14,15);
}

__device__ __forceinline__ float convrelu(const float* xr, float w0, float w1, float w2,
                                          float bb, int pos) {
    float s = bb + w0 * xr[pos] + w1 * xr[pos + 1] + w2 * xr[pos + 2];
    return s > 0.0f ? s : 0.0f;
}

// ---------------------------------------------------------------------------
// Kernel 0: one-shot conversion of fc1_w [32,1040] f32 -> padded f16 [32,1056].
// Removes all bounds checks from the hot GEMM B-fragment path.
// ---------------------------------------------------------------------------
__global__ __launch_bounds__(256)
void k_cvt_w(const float* __restrict__ fc1_w, _Float16* __restrict__ wf16) {
    int i = blockIdx.x * 256 + threadIdx.x;
    if (i >= FC1N * FCKP) return;
    int n = i / FCKP, k = i % FCKP;
    wf16[i] = (k < FCK) ? (_Float16)fc1_w[n * FCK + k] : (_Float16)0.0f;
}

// ---------------------------------------------------------------------------
// Kernel 1: grouped conv1d + relu + maxpool + FC1 GEMM (WMMA), fused.
// 1024 blocks x 128 threads; each block produces 16 rows of z[16384, 32].
// ---------------------------------------------------------------------------
__global__ __launch_bounds__(128)
void k_conv_fc1(const float* __restrict__ x, const float* __restrict__ conv_w,
                const float* __restrict__ conv_b, const _Float16* __restrict__ wf16,
                const float* __restrict__ fc1_b, float* __restrict__ z) {
    __shared__ __align__(16) float xs[16][130];          // padded input rows
    __shared__ __align__(32) _Float16 At[16][FCKP];      // pooled activations, f16
    __shared__ __align__(16) float red[2][256];          // k-split reduction

    const int tid = threadIdx.x;
    const int R0  = blockIdx.x * 16;                     // first flat (b*WIN+w) row

    // stage padded x rows
    for (int i = tid; i < 16 * 130; i += 128) {
        int r = i / 130, p = i % 130;
        float v = 0.0f;
        if (p >= 1 && p <= 128) v = x[(R0 + r) * NFEAT + (p - 1)];
        xs[r][p] = v;
    }
    __syncthreads();

    // conv + relu + maxpool(k2 s2 p1) -> At[r][kc*65+j]
    for (int idx = tid; idx < 16 * FCK; idx += 128) {
        int r = idx / FCK, i = idx % FCK;
        int kc = i / POOL, j = i % POOL;
        int w  = (R0 + r) & (WIN - 1);
        int ch = w * KC + kc;
        float w0 = conv_w[ch * 3 + 0], w1 = conv_w[ch * 3 + 1], w2 = conv_w[ch * 3 + 2];
        float bb = conv_b[ch];
        const float* xr = &xs[r][0];
        float m;
        if (j == 0)        m = convrelu(xr, w0, w1, w2, bb, 0);
        else if (j == 64)  m = convrelu(xr, w0, w1, w2, bb, 127);
        else               m = fmaxf(convrelu(xr, w0, w1, w2, bb, 2 * j - 1),
                                     convrelu(xr, w0, w1, w2, bb, 2 * j));
        At[r][i] = (_Float16)m;
    }
    for (int i = tid; i < 16 * (FCKP - FCK); i += 128)
        At[i / (FCKP - FCK)][FCK + (i % (FCKP - FCK))] = (_Float16)0.0f;
    __syncthreads();

    // WMMA GEMM: [16 x 1056] x [1056 x 32] ; wave -> (ntile, k-half)
    const int wave = tid >> 5, L = tid & 31;
    const bool lob = L < 16;
    const int col = L & 15;
    const int nt  = wave & 1, kh = wave >> 1;
    const int kaoff = lob ? 0 : 16;
    const int kboff = lob ? 0 : 16;
    const int n = nt * 16 + col;

    v8f acc = {};
    const int ks0 = kh ? 17 : 0, ks1 = kh ? 33 : 17;
    for (int ks = ks0; ks < ks1; ++ks) {
        int kb = ks * 32;
        const _Float16* pa = &At[col][kb + kaoff];
        v16h afr = cat8(*(const v8h*)pa, *(const v8h*)(pa + 8));
        const _Float16* pb = wf16 + n * FCKP + kb + kboff;   // 16 contiguous f16
        v16h bfr = cat8(*(const v8h*)pb, *(const v8h*)(pb + 8));
        acc = __builtin_amdgcn_wmma_f32_16x16x32_f16(false, afr, false, bfr,
                                                     (short)0, acc, false, false);
    }
    if (kh == 1) {
        #pragma unroll
        for (int v = 0; v < 8; ++v) red[nt][v * 32 + L] = acc[v];
    }
    __syncthreads();
    if (kh == 0) {
        float bb = fc1_b[n];
        #pragma unroll
        for (int v = 0; v < 8; ++v) {
            int M = v + (lob ? 0 : 8);
            z[(R0 + M) * FC1N + n] = acc[v] + red[nt][v * 32 + L] + bb;
        }
    }
}

// ---------------------------------------------------------------------------
// Kernel 2: 32 independent 2-layer LSTMs, 128 steps, batch 128.
// grid = 32 features x 8 batch-slices = 256 single-wave blocks.
// Per step/wave: 24 x v_wmma_f32_16x16x32_f16; cell state c lives in VGPRs,
// h ping-pongs in wave-private LDS => no barriers inside the time loop.
// ---------------------------------------------------------------------------
__global__ __launch_bounds__(32)
void k_lstm(const float* __restrict__ z, const float* __restrict__ w_ih0,
            const float* __restrict__ w_ih1, const float* __restrict__ w_hh,
            const float* __restrict__ b_lstm, float* __restrict__ hfin) {
    __shared__ __align__(32) _Float16 Bhh0[G4H * HID];
    __shared__ __align__(32) _Float16 Bih1[G4H * HID];
    __shared__ __align__(32) _Float16 Bhh1[G4H * HID];
    __shared__ float wih0s[G4H], b0s[G4H], b1s[G4H];
    __shared__ __align__(32) _Float16 h0s[2][16 * HID];
    __shared__ __align__(32) _Float16 h1s[2][16 * HID];

    const int f     = blockIdx.x & 31;
    const int slice = blockIdx.x >> 5;
    const int base  = slice * 16;
    const int tid   = threadIdx.x;

    for (int i = tid; i < G4H * HID; i += 32) {
        Bhh0[i] = (_Float16)w_hh[f * 4096 + i];          // layer0 w_hh[f]
        Bhh1[i] = (_Float16)w_hh[(32 + f) * 4096 + i];   // layer1 w_hh[f]
        Bih1[i] = (_Float16)w_ih1[f * 4096 + i];         // layer1 w_ih[f]
    }
    for (int i = tid; i < G4H; i += 32) {
        wih0s[i] = w_ih0[f * G4H + i];
        b0s[i]   = b_lstm[f * G4H + i];
        b1s[i]   = b_lstm[(32 + f) * G4H + i];
    }
    for (int i = tid; i < 16 * HID; i += 32) {
        h0s[0][i] = (_Float16)0.0f; h0s[1][i] = (_Float16)0.0f;
        h1s[0][i] = (_Float16)0.0f; h1s[1][i] = (_Float16)0.0f;
    }
    __syncthreads();

    const int  L     = tid;
    const bool lob   = L < 16;
    const int  col   = L & 15;
    const int  kaoff = lob ? 0 : 16;
    const int  kboff = lob ? 0 : 16;
    const int  roff  = lob ? 0 : 8;   // C-fragment row offset

    float c0r[16], c1r[16];
    #pragma unroll
    for (int i = 0; i < 16; ++i) { c0r[i] = 0.0f; c1r[i] = 0.0f; }

    #pragma unroll 1
    for (int t = 0; t < WIN; ++t) {
        const int p = t & 1, q = p ^ 1;

        // x_t column for this wave's batches (rank-1 ih for layer 0)
        float zv[8];
        #pragma unroll
        for (int v = 0; v < 8; ++v)
            zv[v] = z[((base + v + roff) * WIN + t) * FC1N + f];

        // prefetch next step's column (independent of this step's chain)
        if (t + 1 < WIN) {
            #pragma unroll
            for (int v = 0; v < 8; ++v)
                __builtin_prefetch(&z[((base + v + roff) * WIN + t + 1) * FC1N + f], 0, 0);
        }

        // ---- layer 0: gates = h0 x Whh0^T + x_t (x) wih0 + b0 ----
        const _Float16* pa0 = &h0s[p][col * HID + kaoff];
        v16h A0 = cat8(*(const v8h*)pa0, *(const v8h*)(pa0 + 8));

        v8f acc[8];
        #pragma unroll
        for (int nt = 0; nt < 8; ++nt) {
            int g = nt * 16 + col;
            float bb = b0s[g], wz = wih0s[g];
            v8f c;
            #pragma unroll
            for (int v = 0; v < 8; ++v) c[v] = bb + zv[v] * wz;
            v16h Bf = *(const v16h*)&Bhh0[g * HID + kboff];
            acc[nt] = __builtin_amdgcn_wmma_f32_16x16x32_f16(false, A0, false, Bf,
                                                             (short)0, c, false, false);
        }
        #pragma unroll
        for (int hh = 0; hh < 2; ++hh) {
            #pragma unroll
            for (int v = 0; v < 8; ++v) {
                float gi = acc[hh][v],     gf = acc[2 + hh][v];
                float gg = acc[4 + hh][v], go = acc[6 + hh][v];
                float cn = sigf(gf) * c0r[hh * 8 + v] + sigf(gi) * tanhg(gg);
                c0r[hh * 8 + v] = cn;
                float hn = sigf(go) * tanhg(cn);
                h0s[q][(v + roff) * HID + hh * 16 + col] = (_Float16)hn;
            }
        }

        // ---- layer 1: gates = h0_new x Wih1^T + h1 x Whh1^T + b1 ----
        const _Float16* pa1 = &h0s[q][col * HID + kaoff];
        v16h A1 = cat8(*(const v8h*)pa1, *(const v8h*)(pa1 + 8));
        const _Float16* pa2 = &h1s[p][col * HID + kaoff];
        v16h A2 = cat8(*(const v8h*)pa2, *(const v8h*)(pa2 + 8));

        #pragma unroll
        for (int nt = 0; nt < 8; ++nt) {
            int g = nt * 16 + col;
            float bb = b1s[g];
            v8f c;
            #pragma unroll
            for (int v = 0; v < 8; ++v) c[v] = bb;
            v16h Bi = *(const v16h*)&Bih1[g * HID + kboff];
            v16h Bh = *(const v16h*)&Bhh1[g * HID + kboff];
            v8f tmp = __builtin_amdgcn_wmma_f32_16x16x32_f16(false, A1, false, Bi,
                                                             (short)0, c, false, false);
            acc[nt] = __builtin_amdgcn_wmma_f32_16x16x32_f16(false, A2, false, Bh,
                                                             (short)0, tmp, false, false);
        }
        #pragma unroll
        for (int hh = 0; hh < 2; ++hh) {
            #pragma unroll
            for (int v = 0; v < 8; ++v) {
                float gi = acc[hh][v],     gf = acc[2 + hh][v];
                float gg = acc[4 + hh][v], go = acc[6 + hh][v];
                float cn = sigf(gf) * c1r[hh * 8 + v] + sigf(gi) * tanhg(gg);
                c1r[hh * 8 + v] = cn;
                float hn = sigf(go) * tanhg(cn);
                h1s[q][(v + roff) * HID + hh * 16 + col] = (_Float16)hn;
                if (t == WIN - 1)
                    hfin[(base + v + roff) * (FC1N * HID) + f * HID + hh * 16 + col] = hn;
            }
        }
    }
}

// ---------------------------------------------------------------------------
// Kernel 3: mu / log_var heads + interval outputs. 128 blocks x 128 threads.
// out layout: [mu-1.96s | mu | mu+1.96s | log_var], each 128 floats.
// ---------------------------------------------------------------------------
__global__ __launch_bounds__(128)
void k_heads(const float* __restrict__ hfin, const float* __restrict__ mu_w,
             const float* __restrict__ mu_b, const float* __restrict__ lv_w,
             const float* __restrict__ lv_b, float* __restrict__ out) {
    __shared__ float sm[128], sl[128];
    const int b = blockIdx.x, tid = threadIdx.x;
    float pm = 0.0f, pl = 0.0f;
    for (int i = tid; i < FC1N * HID; i += 128) {
        float cm = hfin[b * (FC1N * HID) + i];
        pm += cm * mu_w[i];
        pl += cm * lv_w[i];
    }
    sm[tid] = pm; sl[tid] = pl;
    __syncthreads();
    for (int s = 64; s > 0; s >>= 1) {
        if (tid < s) { sm[tid] += sm[tid + s]; sl[tid] += sl[tid + s]; }
        __syncthreads();
    }
    if (tid == 0) {
        float mu = sm[0] + mu_b[0];
        float lv = sl[0] + lv_b[0];
        float sg = __expf(0.5f * lv);
        out[b]       = mu - 1.96f * sg;
        out[128 + b] = mu;
        out[256 + b] = mu + 1.96f * sg;
        out[384 + b] = lv;
    }
}

extern "C" void kernel_launch(void* const* d_in, const int* in_sizes, int n_in,
                              void* d_out, int out_size, void* d_ws, size_t ws_size,
                              hipStream_t stream) {
    const float* x      = (const float*)d_in[0];
    const float* conv_w = (const float*)d_in[1];
    const float* conv_b = (const float*)d_in[2];
    const float* fc1_w  = (const float*)d_in[3];
    const float* fc1_b  = (const float*)d_in[4];
    const float* w_ih0  = (const float*)d_in[5];
    const float* w_ih1  = (const float*)d_in[6];
    const float* w_hh   = (const float*)d_in[7];
    const float* b_lstm = (const float*)d_in[8];
    const float* mu_w   = (const float*)d_in[9];
    const float* mu_b   = (const float*)d_in[10];
    const float* lv_w   = (const float*)d_in[11];
    const float* lv_b   = (const float*)d_in[12];
    float* out = (float*)d_out;

    float*     zbuf = (float*)d_ws;                        // [16384, 32]
    float*     hbuf = zbuf + (size_t)BATCH * WIN * FC1N;   // [128, 1024]
    _Float16*  wf16 = (_Float16*)(hbuf + BATCH * FC1N * HID); // [32, 1056] f16

    k_cvt_w<<<(FC1N * FCKP + 255) / 256, 256, 0, stream>>>(fc1_w, wf16);
    k_conv_fc1<<<(BATCH * WIN) / 16, 128, 0, stream>>>(x, conv_w, conv_b, wf16, fc1_b, zbuf);
    k_lstm<<<FC1N * 8, 32, 0, stream>>>(zbuf, w_ih0, w_ih1, w_hh, b_lstm, hbuf);
    k_heads<<<BATCH, 128, 0, stream>>>(hbuf, mu_w, mu_b, lv_w, lv_b, out);
}
